// MultiHeadSelfAttentionSubLayer_24764781428980
// MI455X (gfx1250) — matrix operand order repaired and checked by reading, not compiled, not run on hardware
//
#include <hip/hip_runtime.h>

// ---------------------------------------------------------------------------
// Fused attention (flash-style, bf16 WMMA) + output projection for MI455X
// (gfx1250, wave32, WMMA + TDM).  Shapes fixed per reference:
//   Q,K: [16, 2048, 1024] f32   V: [16, 1024, 2048] f32 (pre-transposed)
//   mask:[16, 2048] i32         W: [1024, 1024] f32   b:[1024] f32
//   out: [16, 2048, 1024] f32
//
// Strategy: pre-convert K/V/W to bf16 in d_ws (once, amortized over the 32
// q-tile workgroups per batch that re-read them from L2), then the k-loop
// stages raw bf16 tiles via the Tensor Data Mover (tensor_load_to_lds) and
// all matmuls run on v_wmma_f32_16x16x32_bf16.
// ---------------------------------------------------------------------------

#define MODEL_DIM 1024
#define EMBED_DIM 1024
#define BATCH     16
#define SEQ       2048

#define TQ        64          // query rows per workgroup
#define TK        32          // key columns per k-tile (== WMMA K depth, bf16)
#define NWAVES    8
#define BLOCK     (NWAVES * 32)

typedef __bf16 bf16_t;
typedef __bf16 v16bf __attribute__((ext_vector_type(16)));
typedef __bf16 v8bf  __attribute__((ext_vector_type(8)));
typedef __bf16 v4bf  __attribute__((ext_vector_type(4)));
typedef float  v8f   __attribute__((ext_vector_type(8)));
typedef float  v4f   __attribute__((ext_vector_type(4)));
typedef unsigned int u32x4 __attribute__((ext_vector_type(4)));
typedef int          i32x4 __attribute__((ext_vector_type(4)));
typedef int          i32x8 __attribute__((ext_vector_type(8)));

// LDS layout (bytes) -- total 283392 (< 320 KB / WGP)
#define OFF_Q     0         // 64*1024 bf16 = 131072   (reused for O_bf16)
#define OFF_K     131072    // 32*1024 bf16 =  65536
#define OFF_V     196608    // 1024*32 bf16 =  65536
#define OFF_S     262144    // 64*32 f32    =   8192
#define OFF_P     270336    // 64*32 bf16   =   4096
#define OFF_STATS 274432    // 3*64 f32     =    768
#define OFF_MASK  275200    // 2048 i32     =   8192
#define LDS_TOTAL 283392

// Workspace layout (bytes) for bf16 copies of K, V, W
#define WS_K_OFF  0ull
#define WS_V_OFF  67108864ull                       // 16*2048*1024*2
#define WS_W_OFF  134217728ull
#define WS_NEEDED (134217728ull + 2097152ull)       // + 1024*1024*2

#if defined(__has_builtin)
#if __has_builtin(__builtin_amdgcn_tensor_load_to_lds)
#define HAVE_TDM 1
#endif
#endif

// --- WMMA helper -----------------------------------------------------------
static __device__ __forceinline__ v8f wmma_bf16(v16bf a, v16bf b, v8f c) {
  return __builtin_amdgcn_wmma_f32_16x16x32_bf16(
      /*neg_a=*/false, a, /*neg_b=*/false, b,
      /*c_mod=*/(short)0, c, /*reuse_a=*/false, /*reuse_b=*/false);
}

// A-matrix 16x32 bf16 fragment from a row-major [16 x ld] bf16 tile.
// Layout: lanes 0-15 -> M=lane, K = 0..15;
//         lanes 16-31 -> M=lane-16, K = 16..31 (matches load_frag_b).
static __device__ __forceinline__ v16bf load_frag_a(const bf16_t* base, int ld) {
  const int lane = threadIdx.x & 31;
  const int m    = lane & 15;
  const int kb   = lane & 16;                 // 0 or 16 halves
  const bf16_t* p = base + m * ld + kb;
  v8bf lo = *(const v8bf*)(p);                // K kb..kb+7
  v8bf hi = *(const v8bf*)(p + 8);            // K kb+8..kb+15
  v16bf a;
#pragma unroll
  for (int i = 0; i < 8; ++i) { a[i] = lo[i]; a[i + 8] = hi[i]; }
  return a;
}

// B-matrix 32x16 bf16 fragment: B[k][n] = tile[n][k], tile row-major [16 x ld].
// Lanes 0-15 -> column n=lane, K=0..15; lanes 16-31 -> n=lane-16, K=16..31.
static __device__ __forceinline__ v16bf load_frag_b(const bf16_t* base, int ld) {
  const int lane = threadIdx.x & 31;
  const int n    = lane & 15;
  const int kb   = lane & 16;                 // 0 or 16 halves
  return *(const v16bf*)(base + n * ld + kb);
}

// B fragment with on-the-fly f32->bf16 conversion from global (fallback path).
static __device__ __forceinline__ v16bf load_frag_b_f32(const float* base, int ld) {
  const int lane = threadIdx.x & 31;
  const int n    = lane & 15;
  const int kb   = lane & 16;
  const float* p = base + (size_t)n * ld + kb;
  v16bf b;
#pragma unroll
  for (int i = 0; i < 16; i += 4) {
    v4f f = *(const v4f*)(p + i);
    b[i + 0] = (bf16_t)f.x; b[i + 1] = (bf16_t)f.y;
    b[i + 2] = (bf16_t)f.z; b[i + 3] = (bf16_t)f.w;
  }
  return b;
}

#ifdef HAVE_TDM
// Issue one TDM 2D tile load (global bf16 -> LDS).  D# per CDNA5 ISA §8.3-8.4.
// tensor/tile dims in 2-byte elements; stride0 in elements.
// This toolchain's builtin takes 6 args: (g0 u32x4, g1 i32x8, g2 i32x4,
// g3 i32x4, g4 i32x8, cpol i32); groups 2..4 unused for 2D tiles.
static __device__ __forceinline__ void tdm_load_2d(unsigned lds_off,
                                                   const void* gaddr,
                                                   unsigned tensor_d0,
                                                   unsigned tensor_d1,
                                                   unsigned long long stride0,
                                                   unsigned tile_d0,
                                                   unsigned tile_d1) {
  const unsigned long long ga = (unsigned long long)gaddr;
  u32x4 g0;
  g0[0] = 1u;                                       // count=1, user descriptor
  g0[1] = lds_off;                                  // lds_addr (bytes)
  g0[2] = (unsigned)(ga & 0xffffffffull);           // global_addr[31:0]
  g0[3] = (unsigned)((ga >> 32) & 0x1ffffffull)     // global_addr[56:32]
        | (2u << 30);                               // type=2 ("image")
  i32x8 g1;
  g1[0] = 0x00010000;                               // wg_mask=0, data_size=2B
  g1[1] = (int)((tensor_d0 & 0xffffu) << 16);       // tensor_dim0[15:0]
  g1[2] = (int)((tensor_d0 >> 16) | ((tensor_d1 & 0xffffu) << 16));
  g1[3] = (int)((tensor_d1 >> 16) | (tile_d0 << 16));          // tile_dim0
  g1[4] = (int)(tile_d1 & 0xffffu);                 // tile_dim1, tile_dim2=0
  g1[5] = (int)(stride0 & 0xffffffffull);           // tensor_dim0_stride lo
  g1[6] = (int)((stride0 >> 32) & 0xffffull);       // stride hi; dim1_stride=0
  g1[7] = 0;
  i32x4 z4;
  z4[0] = 0; z4[1] = 0; z4[2] = 0; z4[3] = 0;       // groups 2/3 unused (2D)
  i32x8 z8;
#pragma unroll
  for (int i = 0; i < 8; ++i) z8[i] = 0;
  __builtin_amdgcn_tensor_load_to_lds(g0, g1, z4, z4, z8, 0);
}
#endif

// ---------------------------------------------------------------------------
// f32 -> bf16 pre-conversion (grid-stride, vectorized)
// ---------------------------------------------------------------------------
__global__ void cvt_f32_to_bf16(const float* __restrict__ src,
                                bf16_t* __restrict__ dst, long long n) {
  long long i = ((long long)blockIdx.x * blockDim.x + threadIdx.x) * 4;
  const long long stride = (long long)gridDim.x * blockDim.x * 4;
  for (; i < n; i += stride) {
    v4f f = *(const v4f*)(src + i);
    v4bf o;
    o[0] = (bf16_t)f.x; o[1] = (bf16_t)f.y;
    o[2] = (bf16_t)f.z; o[3] = (bf16_t)f.w;
    *(v4bf*)(dst + i) = o;
  }
}

// ---------------------------------------------------------------------------
// Main fused kernel.  WS=true: K/V/W already bf16 in workspace (TDM staging);
// WS=false: convert f32 tiles on the fly (fallback when ws too small).
// ---------------------------------------------------------------------------
template <bool WS>
__launch_bounds__(BLOCK)
__global__ void mhsa_fused_kernel(const float* __restrict__ Q,
                                  const float* __restrict__ Kf,
                                  const float* __restrict__ Vf,
                                  const int*   __restrict__ mask,
                                  const float* __restrict__ Wf,
                                  const float* __restrict__ bias,
                                  const bf16_t* __restrict__ Kb,
                                  const bf16_t* __restrict__ Vb,
                                  const bf16_t* __restrict__ Wb,
                                  float*       __restrict__ out) {
  extern __shared__ char smem[];
  bf16_t* Qs   = (bf16_t*)(smem + OFF_Q);     // [64][1024], later O_bf16
  bf16_t* Ks   = (bf16_t*)(smem + OFF_K);     // [32][1024]
  bf16_t* Vs   = (bf16_t*)(smem + OFF_V);     // [1024][32]  (d-major)
  float*  Ssc  = (float*) (smem + OFF_S);     // [64][32]
  bf16_t* Ps   = (bf16_t*)(smem + OFF_P);     // [64][32]
  float*  mrow = (float*) (smem + OFF_STATS); // running max [64]
  float*  lrow = mrow + TQ;                   // running sum [64]
  float*  arow = lrow + TQ;                   // rescale     [64]
  int*    mks  = (int*)  (smem + OFF_MASK);   // mask row [2048]

  const int b    = blockIdx.y;
  const int q0   = blockIdx.x * TQ;
  const int tid  = threadIdx.x;
  const int lane = tid & 31;
  const int wave = tid >> 5;
  const int qs    = (wave & 3) * 16;          // wave's q-subtile base
  const int khalf = wave >> 2;                // phase A: k-subtile / B: d-half
  const int rbase = (lane >> 4) << 3;         // C/D row offset (0 or 8)

  // ---- stage Q tile (f32 -> bf16) and mask row into LDS -------------------
  const float* Qg = Q + ((size_t)b * SEQ + q0) * MODEL_DIM;
  for (int i = tid * 4; i < TQ * MODEL_DIM; i += BLOCK * 4) {
    v4f f = *(const v4f*)(Qg + i);
    Qs[i + 0] = (bf16_t)f.x; Qs[i + 1] = (bf16_t)f.y;
    Qs[i + 2] = (bf16_t)f.z; Qs[i + 3] = (bf16_t)f.w;
  }
  for (int i = tid; i < SEQ; i += BLOCK) mks[i] = mask[(size_t)b * SEQ + i];
  if (tid < TQ) { mrow[tid] = -3.0e38f; lrow[tid] = 0.0f; arow[tid] = 1.0f; }

  // O accumulator: 16 q-rows x 512 d-cols per wave = 32 16x16 f32 fragments
  v8f o[32];
#pragma unroll
  for (int f = 0; f < 32; ++f) o[f] = (v8f)0.0f;

  __syncthreads();

  const float scale = 0.03125f;               // 1/sqrt(1024)

  const bf16_t* KbB = WS ? (Kb + (size_t)b * SEQ * MODEL_DIM) : nullptr;
  const bf16_t* VbB = WS ? (Vb + (size_t)b * MODEL_DIM * SEQ) : nullptr;

  for (int kt = 0; kt < SEQ / TK; ++kt) {
    const int k0 = kt * TK;

    // ---- stage K tile [32 x 1024] and V^T tile [1024 x 32] (bf16) ---------
    if constexpr (WS) {
#ifdef HAVE_TDM
      if (wave == 0) {
        // K tile: contiguous 64KB block -> 1-row descriptor
        tdm_load_2d(OFF_K, KbB + (size_t)k0 * MODEL_DIM,
                    TK * MODEL_DIM, 1, TK * MODEL_DIM, TK * MODEL_DIM, 1);
        // V tile: 1024 rows of 32 elems, row stride SEQ elems
        tdm_load_2d(OFF_V, VbB + k0, TK, MODEL_DIM, SEQ, TK, MODEL_DIM);
        __builtin_amdgcn_s_wait_tensorcnt(0);
      }
#else
      const bf16_t* Kt = KbB + (size_t)k0 * MODEL_DIM;
      for (int i = tid * 8; i < TK * MODEL_DIM; i += BLOCK * 8)
        *(v8bf*)(Ks + i) = *(const v8bf*)(Kt + i);
      const bf16_t* Vt = VbB + k0;
      for (int i = tid * 8; i < MODEL_DIM * TK; i += BLOCK * 8) {
        const int d = i >> 5, kk = i & 31;
        *(v8bf*)(Vs + i) = *(const v8bf*)(Vt + (size_t)d * SEQ + kk);
      }
#endif
      // prefetch next tile into L2 (global_prefetch_b8)
      if (kt + 1 < SEQ / TK) {
        const bf16_t* Kn = KbB + (size_t)(k0 + TK) * MODEL_DIM;
        __builtin_prefetch(Kn + (size_t)tid * 128, 0, 1);
        const bf16_t* Vn = VbB + (k0 + TK);
#pragma unroll
        for (int j = 0; j < 4; ++j)
          __builtin_prefetch(Vn + (size_t)(tid * 4 + j) * SEQ, 0, 1);
      }
    } else {
      const float* Kg = Kf + ((size_t)b * SEQ + k0) * MODEL_DIM;
      for (int i = tid * 4; i < TK * MODEL_DIM; i += BLOCK * 4) {
        v4f f = *(const v4f*)(Kg + i);
        Ks[i + 0] = (bf16_t)f.x; Ks[i + 1] = (bf16_t)f.y;
        Ks[i + 2] = (bf16_t)f.z; Ks[i + 3] = (bf16_t)f.w;
      }
      const float* Vg = Vf + (size_t)b * MODEL_DIM * SEQ + k0;
      for (int i = tid * 4; i < MODEL_DIM * TK; i += BLOCK * 4) {
        const int d = i >> 5, kk = i & 31;
        v4f f = *(const v4f*)(Vg + (size_t)d * SEQ + kk);
        bf16_t* dst = Vs + d * TK + kk;
        dst[0] = (bf16_t)f.x; dst[1] = (bf16_t)f.y;
        dst[2] = (bf16_t)f.z; dst[3] = (bf16_t)f.w;
      }
    }
    __syncthreads();

    // ---- phase A: S fragment (16q x 16k), K-depth 1024 --------------------
    {
      v8f c = (v8f)0.0f;
#pragma unroll 8
      for (int dk = 0; dk < MODEL_DIM / 32; ++dk) {
        v16bf a  = load_frag_a(Qs + qs * MODEL_DIM + dk * 32, MODEL_DIM);
        v16bf bb = load_frag_b(Ks + (khalf * 16) * MODEL_DIM + dk * 32, MODEL_DIM);
        c = wmma_bf16(a, bb, c);
      }
      const int n    = lane & 15;
      const int key  = k0 + khalf * 16 + n;
      const bool live = (mks[key] != 0);
#pragma unroll
      for (int r = 0; r < 8; ++r) {
        const int m = qs + r + rbase;
        const float s = c[r] * scale + 1e-13f;           // faithful to ref
        Ssc[m * TK + khalf * 16 + n] = live ? s : -1e-13f;
      }
    }
    __syncthreads();

    // ---- online softmax update (one thread per query row) -----------------
    if (tid < TQ) {
      const float mo = mrow[tid];
      float tm = mo;
#pragma unroll
      for (int j = 0; j < TK; ++j) tm = fmaxf(tm, Ssc[tid * TK + j]);
      const float al = __expf(mo - tm);
      float sum = 0.0f;
#pragma unroll
      for (int j = 0; j < TK; ++j) {
        const float p = __expf(Ssc[tid * TK + j] - tm);
        sum += p;
        Ps[tid * TK + j] = (bf16_t)p;
      }
      mrow[tid] = tm;
      lrow[tid] = lrow[tid] * al + sum;
      arow[tid] = al;
    }
    __syncthreads();

    // ---- phase B: O[16q x 512d] += P[16 x 32] * V^T[32 x 512] -------------
    {
      float al[8];
#pragma unroll
      for (int r = 0; r < 8; ++r) al[r] = arow[qs + r + rbase];
      v16bf a = load_frag_a(Ps + qs * TK, TK);           // reused 32x
#pragma unroll
      for (int f = 0; f < 32; ++f) {
        const int d0 = khalf * 512 + f * 16;
        v16bf bb = load_frag_b(Vs + d0 * TK, TK);
#pragma unroll
        for (int r = 0; r < 8; ++r) o[f][r] *= al[r];    // online rescale
        o[f] = wmma_bf16(a, bb, o[f]);
      }
    }
    __syncthreads();   // protect K/V/S/P tiles before next iteration
  }

  // ---- epilogue: normalize O, store bf16 into Qs region (Q now dead) ------
  {
    float inv[8];
#pragma unroll
    for (int r = 0; r < 8; ++r) inv[r] = 1.0f / lrow[qs + r + rbase];
    const int n = lane & 15;
#pragma unroll
    for (int f = 0; f < 32; ++f) {
      const int d0 = khalf * 512 + f * 16;
#pragma unroll
      for (int r = 0; r < 8; ++r) {
        const int m = qs + r + rbase;
        Qs[m * MODEL_DIM + d0 + n] = (bf16_t)(o[f][r] * inv[r]);
      }
    }
  }
  __syncthreads();

  // ---- projection: out[64 x 1024] = O_bf16[64 x 1024] @ W^T + bias --------
  float* outg = out + ((size_t)b * SEQ + q0) * EMBED_DIM;
  const int e0w = wave * 128;
  for (int qq = 0; qq < 4; ++qq) {
    v8f acc[8];
#pragma unroll
    for (int ef = 0; ef < 8; ++ef) acc[ef] = (v8f)0.0f;
#pragma unroll 4
    for (int dk = 0; dk < MODEL_DIM / 32; ++dk) {
      v16bf a = load_frag_a(Qs + (qq * 16) * MODEL_DIM + dk * 32, MODEL_DIM);
#pragma unroll
      for (int ef = 0; ef < 8; ++ef) {
        v16bf bb;
        if constexpr (WS)
          bb = load_frag_b(Wb + (size_t)(e0w + ef * 16) * MODEL_DIM + dk * 32,
                           MODEL_DIM);
        else
          bb = load_frag_b_f32(Wf + (size_t)(e0w + ef * 16) * MODEL_DIM + dk * 32,
                               MODEL_DIM);
        acc[ef] = wmma_bf16(a, bb, acc[ef]);
      }
    }
    const int n = lane & 15;
#pragma unroll
    for (int ef = 0; ef < 8; ++ef) {
      const int e  = e0w + ef * 16 + n;
      const float bv = bias[e];
#pragma unroll
      for (int r = 0; r < 8; ++r) {
        const int m = qq * 16 + r + rbase;
        outg[(size_t)m * EMBED_DIM + e] = acc[ef][r] + bv;
      }
    }
  }
}

// ---------------------------------------------------------------------------
extern "C" void kernel_launch(void* const* d_in, const int* in_sizes, int n_in,
                              void* d_out, int out_size, void* d_ws, size_t ws_size,
                              hipStream_t stream) {
  (void)in_sizes; (void)n_in; (void)out_size;
  const float* Q    = (const float*)d_in[0];
  const float* Kf   = (const float*)d_in[1];
  const float* Vf   = (const float*)d_in[2];
  const int*   mask = (const int*)  d_in[3];
  const float* Wf   = (const float*)d_in[4];
  const float* bias = (const float*)d_in[5];
  float* out = (float*)d_out;

  dim3 grid(SEQ / TQ, BATCH);   // (32, 16)
  dim3 block(BLOCK);            // 256 threads = 8 waves (wave32)

  if (ws_size >= WS_NEEDED && d_ws != nullptr) {
    bf16_t* Kb = (bf16_t*)((char*)d_ws + WS_K_OFF);
    bf16_t* Vb = (bf16_t*)((char*)d_ws + WS_V_OFF);
    bf16_t* Wb = (bf16_t*)((char*)d_ws + WS_W_OFF);
    const long long nKV = (long long)BATCH * SEQ * MODEL_DIM;
    cvt_f32_to_bf16<<<4096, 256, 0, stream>>>(Kf, Kb, nKV);
    cvt_f32_to_bf16<<<4096, 256, 0, stream>>>(Vf, Vb, nKV);
    cvt_f32_to_bf16<<<1024, 256, 0, stream>>>(Wf, Wb,
                                              (long long)EMBED_DIM * MODEL_DIM);
    mhsa_fused_kernel<true><<<grid, block, LDS_TOTAL, stream>>>(
        Q, Kf, Vf, mask, Wf, bias, Kb, Vb, Wb, out);
  } else {
    mhsa_fused_kernel<false><<<grid, block, LDS_TOTAL, stream>>>(
        Q, Kf, Vf, mask, Wf, bias, nullptr, nullptr, nullptr, out);
  }
}
